// GeoAligner_11209864642634
// MI455X (gfx1250) — matrix operand order repaired and checked by reading, not compiled, not run on hardware
//
#include <hip/hip_runtime.h>
#include <hip/hip_bf16.h>

// ---------- types ----------
typedef __attribute__((ext_vector_type(16))) __bf16        v16bf;
typedef __attribute__((ext_vector_type(8)))  float         v8f;
typedef __attribute__((ext_vector_type(4)))  unsigned int  u32x4;

union FragBF { u32x4 q[2]; v16bf v; };

// ---------- problem constants ----------
#define BSZ     256
#define NIMG    49
#define NIMG_P  64      // padded rows for tile math
#define NGPS    32
#define GDIM    256
#define LROW    264     // LDS row stride in bf16 elems (256 + 8 pad -> conflict-free ds_load_b128)

// ---------- workspace layout (bytes) ----------
#define OFF_IMGP  0u                          // 256*64*256 bf16  = 8 MB
#define OFF_GPSB  (8388608u)                  // 256*32*256 bf16  = 4 MB
#define OFF_L1    (8388608u + 4194304u)       // 256*256 f32
#define OFF_L2    (OFF_L1 + 262144u)          // 256*256 f32
#define OFF_CE    (OFF_L2 + 262144u)          // 512 f32

__device__ __forceinline__ unsigned int f2bf(float f) {
    unsigned int u = __float_as_uint(f);
    return (u + 0x7FFFu + ((u >> 16) & 1u)) >> 16;   // round-to-nearest-even
}

// =====================================================================
// Kernel 1: per-token L2 normalize (fp32), convert to bf16, zero-pad rows
// one wave (32 lanes) per output row; lane handles 8 contiguous dims
// =====================================================================
__global__ __launch_bounds__(256) void norm_bf16_kernel(
    const float* __restrict__ src, unsigned short* __restrict__ dst,
    int rows_src, int rows_dst)
{
    int tid  = blockIdx.x * blockDim.x + threadIdx.x;
    int w    = tid >> 5, lane = tid & 31;
    int batch = w / rows_dst, row = w % rows_dst;
    if (batch >= BSZ) return;

    u32x4* dq = reinterpret_cast<u32x4*>(dst + (size_t)(batch * rows_dst + row) * GDIM);
    if (row < rows_src) {
        const float4* sq = reinterpret_cast<const float4*>(
            src + (size_t)(batch * rows_src + row) * GDIM);
        float4 f0 = sq[lane * 2], f1 = sq[lane * 2 + 1];
        float ss = f0.x*f0.x + f0.y*f0.y + f0.z*f0.z + f0.w*f0.w
                 + f1.x*f1.x + f1.y*f1.y + f1.z*f1.z + f1.w*f1.w;
        #pragma unroll
        for (int m = 1; m < 32; m <<= 1) ss += __shfl_xor(ss, m, 32);
        float inv = 1.0f / fmaxf(sqrtf(ss), 1e-12f);
        u32x4 o;
        o[0] = f2bf(f0.x*inv) | (f2bf(f0.y*inv) << 16);
        o[1] = f2bf(f0.z*inv) | (f2bf(f0.w*inv) << 16);
        o[2] = f2bf(f1.x*inv) | (f2bf(f1.y*inv) << 16);
        o[3] = f2bf(f1.z*inv) | (f2bf(f1.w*inv) << 16);
        dq[lane] = o;
    } else {
        u32x4 z; z[0]=0u; z[1]=0u; z[2]=0u; z[3]=0u;
        dq[lane] = z;                        // zero padding rows 49..63
    }
}

// =====================================================================
// Kernel 2: per (b,g) pair: S = img_b(64x256) x gps_g^T(256x32) via
// v_wmma_f32_16x16x32_bf16, then rowmax-mean and colmax-mean reductions.
// grid = (256 batches, 32 g-blocks), block = 256 threads (8 waves),
// wave w handles g = blockIdx.y*8 + w. img_b staged in LDS (padded rows).
// =====================================================================
__global__ __launch_bounds__(256) void pair_kernel(
    const unsigned short* __restrict__ imgp,
    const unsigned short* __restrict__ gpsb,
    const float* __restrict__ logit_scale,
    float* __restrict__ L1, float* __restrict__ L2)
{
    __shared__ __align__(16) unsigned short simg[NIMG_P * LROW];

    const int b   = blockIdx.x;
    const int tid = threadIdx.x;

    // cooperative copy: 64 rows x 512B (32 x 16B chunks per row), padded rows in LDS
    {
        const u32x4* g4 = reinterpret_cast<const u32x4*>(imgp + (size_t)b * NIMG_P * GDIM);
        #pragma unroll
        for (int i = 0; i < 8; ++i) {
            int c = tid + i * 256;            // chunk id 0..2047
            int row = c >> 5, off = c & 31;   // off in 16B units
            u32x4 v = g4[c];
            *reinterpret_cast<u32x4*>(&simg[row * LROW + off * 8]) = v;
        }
    }
    __syncthreads();

    const int wave = tid >> 5, lane = tid & 31;
    const int g    = blockIdx.y * 8 + wave;
    const int half = lane >> 4, l15 = lane & 15;

    v8f acc[4][2];
    #pragma unroll
    for (int mt = 0; mt < 4; ++mt)
        #pragma unroll
        for (int nt = 0; nt < 2; ++nt)
            #pragma unroll
            for (int v = 0; v < 8; ++v) acc[mt][nt][v] = 0.0f;

    // B fragment row pointers (gps token rows, read from global/L2)
    const u32x4* brow[2];
    #pragma unroll
    for (int nt = 0; nt < 2; ++nt)
        brow[nt] = reinterpret_cast<const u32x4*>(
            gpsb + ((size_t)g * NGPS + nt * 16 + l15) * GDIM);

    #pragma unroll
    for (int k0 = 0; k0 < 8; ++k0) {          // K = 256 in steps of 32
        // B 32x16 bf16 (N=l15): lanes<16: v0-3=K+0..7, v4-7=K+16..23 ; lanes>=16: +8
        FragBF bf[2];
        #pragma unroll
        for (int nt = 0; nt < 2; ++nt) {
            bf[nt].q[0] = brow[nt][k0 * 4 + half];
            bf[nt].q[1] = brow[nt][k0 * 4 + 2 + half];
        }
        #pragma unroll
        for (int mt = 0; mt < 4; ++mt) {
            // A 16x32 bf16: lanes<16: v0-3=K+0..7, v4-7=K+16..23 ; lanes>=16: +8
            FragBF a;
            const unsigned short* ar = &simg[(mt * 16 + l15) * LROW + k0 * 32 + 8 * half];
            a.q[0] = *reinterpret_cast<const u32x4*>(ar);
            a.q[1] = *reinterpret_cast<const u32x4*>(ar + 16);
            #pragma unroll
            for (int nt = 0; nt < 2; ++nt)
                acc[mt][nt] = __builtin_amdgcn_wmma_f32_16x16x32_bf16(
                    false, a.v, false, bf[nt].v, (short)0, acc[mt][nt], false, false);
        }
    }

    const float scale = fminf(expf(logit_scale[0]), 100.0f);

    // ---- logits_i2g[b,g] = scale * mean_{n<49} max_m S[n,m] ----
    // D layout: lane holds S[n = mt*16 + v + 8*half, m = nt*16 + l15] in acc[mt][nt][v]
    float rsum = 0.0f;
    #pragma unroll
    for (int mt = 0; mt < 4; ++mt) {
        #pragma unroll
        for (int v = 0; v < 8; ++v) {
            float t = fmaxf(acc[mt][0][v], acc[mt][1][v]);
            t = fmaxf(t, __shfl_xor(t, 1, 32));
            t = fmaxf(t, __shfl_xor(t, 2, 32));
            t = fmaxf(t, __shfl_xor(t, 4, 32));
            t = fmaxf(t, __shfl_xor(t, 8, 32));   // max over 16 m within half
            int n = mt * 16 + v + 8 * half;
            if (n < NIMG) rsum += t;              // exclude zero-padded rows
        }
    }
    rsum += __shfl_xor(rsum, 16, 32);             // combine the two half-wave row sets

    // ---- logits_g2i[g,b] = scale * mean_{m<32} max_{n<49} S[n,m] ----
    float csum = 0.0f;
    #pragma unroll
    for (int nt = 0; nt < 2; ++nt) {
        float c = -3.0e38f;
        #pragma unroll
        for (int mt = 0; mt < 3; ++mt)            // rows 0..47 always valid
            #pragma unroll
            for (int v = 0; v < 8; ++v) c = fmaxf(c, acc[mt][nt][v]);
        if (half == 0) c = fmaxf(c, acc[3][nt][0]);   // row n = 48 only
        c = fmaxf(c, __shfl_xor(c, 16, 32));      // combine n-halves for same m
        csum += c;
    }
    csum += __shfl_xor(csum, 1, 32);
    csum += __shfl_xor(csum, 2, 32);
    csum += __shfl_xor(csum, 4, 32);
    csum += __shfl_xor(csum, 8, 32);              // sum over 16 lanes -> all 32 m

    if (lane == 0) {
        L1[b * BSZ + g] = rsum * (scale / (float)NIMG);
        L2[g * BSZ + b] = csum * (scale / (float)NGPS);
    }
}

// =====================================================================
// Kernel 3a: per-row cross-entropy term (stable logsumexp). One wave/row,
// 512 rows (256 of L1 then 256 of L2), lane holds 8 contiguous logits.
// =====================================================================
__global__ __launch_bounds__(256) void ce_rows_kernel(
    const float* __restrict__ L1, const float* __restrict__ L2,
    float* __restrict__ ce)
{
    int tid = blockIdx.x * blockDim.x + threadIdx.x;
    int row = tid >> 5, lane = tid & 31;
    const float* Lg = (row < BSZ) ? (L1 + (size_t)row * BSZ)
                                  : (L2 + (size_t)(row - BSZ) * BSZ);
    int label = row & (BSZ - 1);
    const float4* q = reinterpret_cast<const float4*>(Lg);
    float4 a = q[lane * 2], b4 = q[lane * 2 + 1];
    float m = fmaxf(fmaxf(fmaxf(a.x, a.y), fmaxf(a.z, a.w)),
                    fmaxf(fmaxf(b4.x, b4.y), fmaxf(b4.z, b4.w)));
    #pragma unroll
    for (int s = 1; s < 32; s <<= 1) m = fmaxf(m, __shfl_xor(m, s, 32));
    float e = expf(a.x - m) + expf(a.y - m) + expf(a.z - m) + expf(a.w - m)
            + expf(b4.x - m) + expf(b4.y - m) + expf(b4.z - m) + expf(b4.w - m);
    #pragma unroll
    for (int s = 1; s < 32; s <<= 1) e += __shfl_xor(e, s, 32);
    if (lane == 0) ce[row] = m + logf(e) - Lg[label];
}

// =====================================================================
// Kernel 3b: reduce 512 CE terms -> loss = (mean(CE1)+mean(CE2))/2
// =====================================================================
__global__ __launch_bounds__(256) void final_kernel(
    const float* __restrict__ ce, float* __restrict__ out)
{
    __shared__ float sh[8];
    int tid = threadIdx.x, lane = tid & 31, w = tid >> 5;
    float s = ce[tid] + ce[tid + 256];
    #pragma unroll
    for (int m = 1; m < 32; m <<= 1) s += __shfl_xor(s, m, 32);
    if (lane == 0) sh[w] = s;
    __syncthreads();
    if (tid == 0) {
        float t = 0.0f;
        #pragma unroll
        for (int i = 0; i < 8; ++i) t += sh[i];
        out[0] = t / 512.0f;   // ( sum1/256 + sum2/256 ) / 2
    }
}

// =====================================================================
extern "C" void kernel_launch(void* const* d_in, const int* in_sizes, int n_in,
                              void* d_out, int out_size, void* d_ws, size_t ws_size,
                              hipStream_t stream)
{
    const float* img = (const float*)d_in[0];   // (256,49,256) f32
    const float* gps = (const float*)d_in[1];   // (256,32,256) f32
    const float* ls  = (const float*)d_in[2];   // scalar f32

    char* ws = (char*)d_ws;
    unsigned short* imgp = (unsigned short*)(ws + OFF_IMGP);
    unsigned short* gpsb = (unsigned short*)(ws + OFF_GPSB);
    float* L1 = (float*)(ws + OFF_L1);
    float* L2 = (float*)(ws + OFF_L2);
    float* ce = (float*)(ws + OFF_CE);

    // normalize + bf16 convert (+pad img rows 49->64 with zeros)
    norm_bf16_kernel<<<dim3(2048), dim3(256), 0, stream>>>(img, imgp, NIMG, NIMG_P);
    norm_bf16_kernel<<<dim3(1024), dim3(256), 0, stream>>>(gps, gpsb, NGPS, NGPS);

    // all-pairs max-sim logits via WMMA
    pair_kernel<<<dim3(256, 32), dim3(256), 0, stream>>>(imgp, gpsb, ls, L1, L2);

    // cross-entropy both directions + final scalar loss
    ce_rows_kernel<<<dim3(64), dim3(256), 0, stream>>>(L1, L2, ce);
    final_kernel<<<dim3(1), dim3(256), 0, stream>>>(ce, (float*)d_out);
}
